// AttentionBlock_27350351740953
// MI455X (gfx1250) — matrix operand order repaired and checked by reading, not compiled, not run on hardware
//
#include <hip/hip_runtime.h>
#include <hip/hip_bf16.h>

typedef __bf16 bf16_t;
typedef __attribute__((ext_vector_type(16))) __bf16 v16bf;
typedef __attribute__((ext_vector_type(8)))  __bf16 v8bf;
typedef __attribute__((ext_vector_type(8)))  float  v8f;
typedef int v4i __attribute__((vector_size(16)));

#define B_SZ 8
#define C_SZ 512
#define T_SZ 1024
#define NH   8
#define CH   64          // C_SZ / NH
#define NG   32
#define CPG  16          // C_SZ / NG
#define EPS_GN 1e-5f
#define QK_SCALE 0.35355339059327373f   // 1 / sqrt(sqrt(64))

// LDS row stride (bf16 elements): rows 16B-aligned, r*stride conflict-free
// across the 16 lanes of a half-wave.
#define LDW64 72   // 64-wide tiles -> 144 B rows

// ---------------------------------------------------------------------------
// CDNA5 async global->LDS copy (ASYNCcnt path); uint4 fallback.
// ---------------------------------------------------------------------------
#if __has_builtin(__builtin_amdgcn_global_load_async_to_lds_b128)
#define HAVE_ASYNC_LDS 1
typedef __attribute__((address_space(1))) v4i* gas_v4i_ptr;
typedef __attribute__((address_space(3))) v4i* las_v4i_ptr;
#endif

__device__ __forceinline__ void cp_b128(void* dst_lds, const void* src_g) {
#if defined(HAVE_ASYNC_LDS)
  __builtin_amdgcn_global_load_async_to_lds_b128(
      (gas_v4i_ptr)(src_g), (las_v4i_ptr)(dst_lds), 0, 0);
#else
  *(uint4*)dst_lds = *(const uint4*)src_g;
#endif
}

__device__ __forceinline__ void cp_wait() {
#if defined(HAVE_ASYNC_LDS)
  asm volatile("s_wait_asynccnt 0x0" ::: "memory");
#endif
}

// ---------------------------------------------------------------------------
// WMMA fragment helpers (CDNA5 wave32 layouts, CDNA5 ISA §7.12.2)
// ---------------------------------------------------------------------------

// A fragment: 16x32 bf16 (MxK), row-major source, leading dim lda (elements).
// lane (r,hi): element e -> M = r, K = hi*16 + e
__device__ __forceinline__ v16bf frag_load_A(const bf16_t* base, int lda) {
  const int lane = threadIdx.x & 31;
  const int r = lane & 15, hi = lane >> 4;
  v16bf a;
#pragma unroll
  for (int e = 0; e < 16; ++e) {
    const int k = hi * 16 + e;
    a[e] = base[r * lda + k];
  }
  return a;
}

// B fragment (32x16 KxN) from an N-major (transposed) tile Bt[n][k], ldt.
// lane (r,hi): element e -> K = hi*16 + e, N = r.
__device__ __forceinline__ v16bf frag_load_BT(const bf16_t* base, int ldt) {
  const int lane = threadIdx.x & 31;
  const int r = lane & 15, hi = lane >> 4;
  v16bf b;
#pragma unroll
  for (int e = 0; e < 16; ++e)
    b[e] = base[r * ldt + hi * 16 + e];
  return b;
}

__device__ __forceinline__ v8f wmma_bf16(v16bf a, v16bf b, v8f c) {
  return __builtin_amdgcn_wmma_f32_16x16x32_bf16(
      false, a, false, b, (short)0, c, false, false);
}
// C/D fragment element (vgpr i, lane (r,hi)) -> M = i + hi*8, N = r

// ---------------------------------------------------------------------------
// Kernel 1: GroupNorm, fp32 [B,C,T] -> bf16 xnT [B,T,C]  (t-major!)
// ---------------------------------------------------------------------------
__global__ __launch_bounds__(256) void gn_kernel(const float* __restrict__ x,
                                                 const float* __restrict__ w,
                                                 const float* __restrict__ bias,
                                                 bf16_t* __restrict__ xnT) {
  const int g = blockIdx.x;              // b*NG + grp
  const int b = g >> 5, grp = g & 31;
  const size_t base = ((size_t)b * C_SZ + (size_t)grp * CPG) * T_SZ;
  const float* xg = x + base;
  const int N = CPG * T_SZ;              // 16384 elements per group
  float s = 0.f, s2 = 0.f;
  for (int i = threadIdx.x; i < N; i += 256) {
    const float v = xg[i];
    s += v; s2 += v * v;
  }
#pragma unroll
  for (int off = 16; off > 0; off >>= 1) {
    s  += __shfl_xor(s,  off, 32);
    s2 += __shfl_xor(s2, off, 32);
  }
  __shared__ float rs[8], rs2[8];
  const int wv = threadIdx.x >> 5, ln = threadIdx.x & 31;
  if (ln == 0) { rs[wv] = s; rs2[wv] = s2; }
  __syncthreads();
  if (threadIdx.x == 0) {
    float a = 0.f, a2 = 0.f;
#pragma unroll
    for (int i = 0; i < 8; ++i) { a += rs[i]; a2 += rs2[i]; }
    rs[0] = a; rs2[0] = a2;
  }
  __syncthreads();
  const float mean = rs[0] / (float)N;
  const float var  = rs2[0] / (float)N - mean * mean;
  const float rstd = rsqrtf(var + EPS_GN);

  float gw[CPG], gb[CPG];
#pragma unroll
  for (int cc = 0; cc < CPG; ++cc) {
    gw[cc] = w[grp * CPG + cc] * rstd;
    gb[cc] = bias[grp * CPG + cc];
  }
  for (int t = threadIdx.x; t < T_SZ; t += 256) {
    v8bf lo, hi2;
#pragma unroll
    for (int cc = 0; cc < 8; ++cc)
      lo[cc] = (bf16_t)((xg[cc * T_SZ + t] - mean) * gw[cc] + gb[cc]);
#pragma unroll
    for (int cc = 0; cc < 8; ++cc)
      hi2[cc] = (bf16_t)((xg[(cc + 8) * T_SZ + t] - mean) * gw[cc + 8] + gb[cc + 8]);
    bf16_t* dst = xnT + ((size_t)b * T_SZ + t) * C_SZ + grp * CPG;
    *(v8bf*)(dst)     = lo;
    *(v8bf*)(dst + 8) = hi2;
  }
}

// ---------------------------------------------------------------------------
// Shared GEMM machinery: 64x64 output tile, K-step 64, double-buffered LDS.
// A: fp32 weights [M][C] -> bf16 regs -> LDS.  B: bf16 N-major rows via async.
// ---------------------------------------------------------------------------
#define GEMM_PREP()                                                            \
  const int tid = threadIdx.x, wave = tid >> 5;                                \
  const int arow = tid >> 3;           /* 0..15 */                             \
  const int acol = (tid & 7) * 8;      /* 0..56 */

#define GEMM_LOAD_AREG(W, k0)                                                  \
  _Pragma("unroll") for (int it = 0; it < 4; ++it) {                           \
    const float* s_ = &(W)[(size_t)(m0 + arow + 16 * it) * C_SZ + (k0) + acol];\
    _Pragma("unroll") for (int e = 0; e < 8; ++e) areg[it][e] = s_[e];         \
  }

#define GEMM_STORE_AREG(buf)                                                   \
  _Pragma("unroll") for (int it = 0; it < 4; ++it) {                           \
    v8bf p_;                                                                   \
    _Pragma("unroll") for (int e = 0; e < 8; ++e) p_[e] = (bf16_t)areg[it][e]; \
    *(v8bf*)&As[buf][arow + 16 * it][acol] = p_;                               \
  }

#define GEMM_FILL_B(buf, src /* bf16 base at [n0][k0], row stride C_SZ */)     \
  _Pragma("unroll") for (int it = 0; it < 4; ++it)                             \
    cp_b128(&Bt[buf][arow + 16 * it][acol],                                    \
            &(src)[(size_t)(arow + 16 * it) * C_SZ + acol]);

#define GEMM_COMPUTE(buf)                                                      \
  _Pragma("unroll") for (int kk = 0; kk < 64; kk += 32) {                      \
    const v16bf af = frag_load_A(&As[buf][wave * 16][kk], LDW64);              \
    _Pragma("unroll") for (int nf = 0; nf < 4; ++nf) {                         \
      const v16bf bfb = frag_load_BT(&Bt[buf][nf * 16][kk], LDW64);            \
      acc[nf] = wmma_bf16(af, bfb, acc[nf]);                                   \
    }                                                                          \
  }

// ---------------------------------------------------------------------------
// Kernel 2: QKV GEMM  qkv[b,o,t] = sum_c W[o,c]*xnT[b,t,c] + bias[o]
//   o = hh*192 + part*64 + c   (reshape [B,3C,T] -> [B*NH, 3*ch, T])
//   q -> qbuf[bh][t][c]*scale ; k -> kbuf[bh][s][c]*scale ; v -> vbuf[bh][c][s]
// ---------------------------------------------------------------------------
__global__ __launch_bounds__(128) void qkv_gemm_kernel(
    const float* __restrict__ qkv_w, const float* __restrict__ qkv_b,
    const bf16_t* __restrict__ xnT,
    bf16_t* __restrict__ qbuf, bf16_t* __restrict__ kbuf,
    bf16_t* __restrict__ vbuf) {
  __shared__ bf16_t As[2][64][LDW64];
  __shared__ bf16_t Bt[2][64][LDW64];
  const int m0 = blockIdx.x * 64;
  const int n0 = blockIdx.y * 64;
  const int b  = blockIdx.z;
  GEMM_PREP();
  const bf16_t* xb = xnT + ((size_t)b * T_SZ + n0) * C_SZ;

  v8f acc[4] = {};
  {  // prologue: stage tile 0
    float areg[4][8];
    GEMM_LOAD_AREG(qkv_w, 0);
    GEMM_FILL_B(0, xb);
    GEMM_STORE_AREG(0);
    cp_wait();
    __syncthreads();
  }
  int cur = 0;
  for (int k0 = 0; k0 < C_SZ; k0 += 64) {
    const int nxt = cur ^ 1;
    const bool hn = (k0 + 64) < C_SZ;
    float areg[4][8];
    if (hn) {                      // issue next-tile traffic before compute
      GEMM_LOAD_AREG(qkv_w, k0 + 64);
      GEMM_FILL_B(nxt, xb + (k0 + 64));
    }
    GEMM_COMPUTE(cur);
    if (hn) GEMM_STORE_AREG(nxt);  // global-load wait lands after the WMMAs
    cp_wait();
    __syncthreads();
    cur = nxt;
  }

  const int lane = tid & 31, r = lane & 15, hi = lane >> 4;
#pragma unroll
  for (int nf = 0; nf < 4; ++nf) {
#pragma unroll
    for (int i = 0; i < 8; ++i) {
      const int o = m0 + wave * 16 + i + hi * 8;
      const int t = n0 + nf * 16 + r;
      const float val = acc[nf][i] + qkv_b[o];
      const int hh = o / 192;
      const int rem = o - hh * 192;
      const int part = rem >> 6;
      const int c = rem & 63;
      const int bh = b * NH + hh;
      if (part == 0)
        qbuf[((size_t)bh * T_SZ + t) * CH + c] = (bf16_t)(val * QK_SCALE);
      else if (part == 1)
        kbuf[((size_t)bh * T_SZ + t) * CH + c] = (bf16_t)(val * QK_SCALE);
      else
        vbuf[((size_t)bh * CH + c) * T_SZ + t] = (bf16_t)val;
    }
  }
}

// ---------------------------------------------------------------------------
// Kernel 3: flash attention per (q-tile, batch*head), double-buffered K/V
//   S = Q K^T : A = Qs[t][c], B via Kt[s][c] (N-major)
//   O += P V  : A = Ps[t][s], B via Vt[c][s] (N-major)
// ---------------------------------------------------------------------------
__global__ __launch_bounds__(128) void attn_kernel(
    const bf16_t* __restrict__ qbuf, const bf16_t* __restrict__ kbuf,
    const bf16_t* __restrict__ vbuf, bf16_t* __restrict__ hT) {
  __shared__ bf16_t Qs[64][LDW64];      // [t][c]
  __shared__ bf16_t Kt[2][64][LDW64];   // [s][c]
  __shared__ bf16_t Vt[2][64][LDW64];   // [c][s]
  __shared__ bf16_t Ps[64][LDW64];      // [t][s] wave-private 16-row slices
  const int qb = blockIdx.x;
  const int bh = blockIdx.y;
  const int b = bh >> 3, hh = bh & 7;
  const int tid = threadIdx.x, wave = tid >> 5;
  const int lane = tid & 31, r = lane & 15, hi = lane >> 4;
  const int frow = tid >> 3;            // 0..15
  const int fcol = (tid & 7) * 8;       // 0..56

  const bf16_t* qg = qbuf + ((size_t)bh * T_SZ + qb * 64) * CH;
  const bf16_t* kg0 = kbuf + (size_t)bh * T_SZ * CH;     // [s][c]
  const bf16_t* vg0 = vbuf + (size_t)bh * CH * T_SZ;     // [c][s]

  // prologue: Q tile + K/V tile 0 (all async)
#pragma unroll
  for (int it = 0; it < 4; ++it) {
    const int rr = frow + 16 * it;
    cp_b128(&Qs[rr][fcol], &qg[(size_t)rr * CH + fcol]);
    cp_b128(&Kt[0][rr][fcol], &kg0[(size_t)rr * CH + fcol]);
    cp_b128(&Vt[0][rr][fcol], &vg0[(size_t)rr * T_SZ + fcol]);
  }
  cp_wait();
  __syncthreads();

  float m_run[8], l_run[8];
  v8f oacc[4] = {};
#pragma unroll
  for (int i = 0; i < 8; ++i) { m_run[i] = -3.0e38f; l_run[i] = 0.f; }

  int cur = 0;
  for (int s0 = 0; s0 < T_SZ; s0 += 64) {
    const int nxt = cur ^ 1;
    const bool hn = (s0 + 64) < T_SZ;
    if (hn) {                    // prefetch next K/V tile while computing
#pragma unroll
      for (int it = 0; it < 4; ++it) {
        const int rr = frow + 16 * it;
        cp_b128(&Kt[nxt][rr][fcol], &kg0[(size_t)(s0 + 64 + rr) * CH + fcol]);
        cp_b128(&Vt[nxt][rr][fcol], &vg0[(size_t)rr * T_SZ + s0 + 64 + fcol]);
      }
    }

    // S = Q * K^T over c = 64 (two k-steps)
    v8f sacc[4] = {};
#pragma unroll
    for (int kk = 0; kk < 64; kk += 32) {
      const v16bf af = frag_load_A(&Qs[wave * 16][kk], LDW64);
#pragma unroll
      for (int sf = 0; sf < 4; ++sf) {
        const v16bf bfb = frag_load_BT(&Kt[cur][sf * 16][kk], LDW64);
        sacc[sf] = wmma_bf16(af, bfb, sacc[sf]);
      }
    }

    // online softmax (row M = i + hi*8 lives in the 16 lanes sharing hi)
#pragma unroll
    for (int i = 0; i < 8; ++i) {
      float mx = -3.0e38f;
#pragma unroll
      for (int sf = 0; sf < 4; ++sf) mx = fmaxf(mx, sacc[sf][i]);
#pragma unroll
      for (int off = 1; off < 16; off <<= 1)
        mx = fmaxf(mx, __shfl_xor(mx, off, 32));
      const float m_new = fmaxf(m_run[i], mx);
      const float corr = __expf(m_run[i] - m_new);
      m_run[i] = m_new;
      const int tl = wave * 16 + i + hi * 8;
      float sum = 0.f;
#pragma unroll
      for (int sf = 0; sf < 4; ++sf) {
        const float p = __expf(sacc[sf][i] - m_new);
        sum += p;
        Ps[tl][sf * 16 + r] = (bf16_t)p;
      }
#pragma unroll
      for (int off = 1; off < 16; off <<= 1)
        sum += __shfl_xor(sum, off, 32);
      l_run[i] = l_run[i] * corr + sum;
#pragma unroll
      for (int cf = 0; cf < 4; ++cf) oacc[cf][i] = oacc[cf][i] * corr;
    }

    // O += P * V (Ps rows are wave-private; same-wave DS ops stay in order)
#pragma unroll
    for (int kk = 0; kk < 64; kk += 32) {
      const v16bf af = frag_load_A(&Ps[wave * 16][kk], LDW64);
#pragma unroll
      for (int cf = 0; cf < 4; ++cf) {
        const v16bf bfb = frag_load_BT(&Vt[cur][cf * 16][kk], LDW64);
        oacc[cf] = wmma_bf16(af, bfb, oacc[cf]);
      }
    }
    cp_wait();        // next tile's copies had the whole compute to finish
    __syncthreads();
    cur = nxt;
  }

#pragma unroll
  for (int cf = 0; cf < 4; ++cf) {
#pragma unroll
    for (int i = 0; i < 8; ++i) {
      const int tl = wave * 16 + i + hi * 8;
      const int t = qb * 64 + tl;
      const int c = cf * 16 + r;
      const float o = oacc[cf][i] / l_run[i];
      hT[((size_t)b * T_SZ + t) * C_SZ + hh * CH + c] = (bf16_t)o;
    }
  }
}

// ---------------------------------------------------------------------------
// Kernel 4: proj GEMM + bias + residual -> fp32 out, double-buffered
// ---------------------------------------------------------------------------
__global__ __launch_bounds__(128) void proj_kernel(
    const float* __restrict__ pw, const float* __restrict__ pb,
    const bf16_t* __restrict__ hT, const float* __restrict__ x,
    float* __restrict__ out) {
  __shared__ bf16_t As[2][64][LDW64];
  __shared__ bf16_t Bt[2][64][LDW64];
  const int m0 = blockIdx.x * 64;
  const int n0 = blockIdx.y * 64;
  const int b  = blockIdx.z;
  GEMM_PREP();
  const bf16_t* hb = hT + ((size_t)b * T_SZ + n0) * C_SZ;

  v8f acc[4] = {};
  {
    float areg[4][8];
    GEMM_LOAD_AREG(pw, 0);
    GEMM_FILL_B(0, hb);
    GEMM_STORE_AREG(0);
    cp_wait();
    __syncthreads();
  }
  int cur = 0;
  for (int k0 = 0; k0 < C_SZ; k0 += 64) {
    const int nxt = cur ^ 1;
    const bool hn = (k0 + 64) < C_SZ;
    float areg[4][8];
    if (hn) {
      GEMM_LOAD_AREG(pw, k0 + 64);
      GEMM_FILL_B(nxt, hb + (k0 + 64));
    }
    GEMM_COMPUTE(cur);
    if (hn) GEMM_STORE_AREG(nxt);
    cp_wait();
    __syncthreads();
    cur = nxt;
  }

  const int lane = tid & 31, r = lane & 15, hi = lane >> 4;
#pragma unroll
  for (int nf = 0; nf < 4; ++nf) {
#pragma unroll
    for (int i = 0; i < 8; ++i) {
      const int o = m0 + wave * 16 + i + hi * 8;
      const int t = n0 + nf * 16 + r;
      const size_t off = ((size_t)b * C_SZ + o) * T_SZ + t;
      out[off] = acc[nf][i] + pb[o] + x[off];
    }
  }
}

// ---------------------------------------------------------------------------
extern "C" void kernel_launch(void* const* d_in, const int* in_sizes, int n_in,
                              void* d_out, int out_size, void* d_ws, size_t ws_size,
                              hipStream_t stream) {
  (void)in_sizes; (void)n_in; (void)out_size; (void)ws_size;
  const float* x      = (const float*)d_in[0];
  const float* gn_w   = (const float*)d_in[1];
  const float* gn_b   = (const float*)d_in[2];
  const float* qkv_w  = (const float*)d_in[3];
  const float* qkv_b  = (const float*)d_in[4];
  const float* proj_w = (const float*)d_in[5];
  const float* proj_b = (const float*)d_in[6];
  float* out = (float*)d_out;

  const size_t SEG = (size_t)B_SZ * C_SZ * T_SZ * sizeof(bf16_t);  // 8 MiB
  char* ws = (char*)d_ws;
  bf16_t* xnT  = (bf16_t*)(ws);            // [B][T][C]
  bf16_t* qbuf = (bf16_t*)(ws + SEG);      // [B*NH][T][CH]
  bf16_t* kbuf = (bf16_t*)(ws + 2 * SEG);  // [B*NH][T][CH]
  bf16_t* vbuf = (bf16_t*)(ws + 3 * SEG);  // [B*NH][CH][T]
  bf16_t* hT   = (bf16_t*)(ws + 4 * SEG);  // [B][T][C]

  gn_kernel<<<dim3(B_SZ * NG), dim3(256), 0, stream>>>(x, gn_w, gn_b, xnT);
  qkv_gemm_kernel<<<dim3(24, 16, B_SZ), dim3(128), 0, stream>>>(
      qkv_w, qkv_b, xnT, qbuf, kbuf, vbuf);
  attn_kernel<<<dim3(16, 64), dim3(128), 0, stream>>>(qbuf, kbuf, vbuf, hT);
  proj_kernel<<<dim3(8, 16, B_SZ), dim3(128), 0, stream>>>(
      proj_w, proj_b, hT, x, out);
}
